// KimiK2_2001454759956
// MI455X (gfx1250) — matrix operand order repaired and checked by reading, not compiled, not run on hardware
//
#include <hip/hip_runtime.h>
#include <math.h>
#include <stdint.h>

// ---------------- config (matches reference) ----------------
#define DEPTH 2
#define DIM   2048
#define NH    16
#define HD    128
#define RLAT  512
#define DFF   1024
#define NE    4
#define TOPK  2
#define VOC   32000
#define BB    2
#define SS    1024
#define TT    (BB * SS)   // 2048 tokens

typedef __attribute__((ext_vector_type(16))) __bf16 v16bf;
typedef __attribute__((ext_vector_type(8)))  float  v8f;
typedef unsigned int u32x4 __attribute__((ext_vector_type(4)));
typedef int          i32x8 __attribute__((ext_vector_type(8)));
typedef int          i32x4 __attribute__((ext_vector_type(4)));

__device__ __forceinline__ __bf16 f2bf(float f) { return (__bf16)f; }

__device__ __forceinline__ v8f wmma_bf16(v16bf a, v16bf b, v8f c) {
  return __builtin_amdgcn_wmma_f32_16x16x32_bf16(
      /*neg_a=*/false, a, /*neg_b=*/false, b,
      /*c_mod=*/(short)0, c, /*reuse_a=*/false, /*reuse_b=*/false);
}

// ---- Tensor Data Mover: 2D fp32 tile Global -> LDS (D# per CDNA5 ISA) ----
// group0: [1:0]=count=1, [63:32]=lds_addr, [120:64]=global_addr, [127:126]=type=2
// group1: [17:16]=data_size(2=4B), [20]=pad_en, [24:22]=pad_interval, [31:25]=pad_amt(0->1DW)
//         [79:48]=tensor_dim0, [111:80]=tensor_dim1, [127:112]=tile_dim0,
//         [143:128]=tile_dim1, [207:160]=tensor_dim0_stride
// This toolchain exposes the 6-arg builtin: (g0, g1, g2, g3, g_extra, cpol).
__device__ __forceinline__ void tdm_load_2d_f32(
    const float* gaddr, unsigned lds_off,
    unsigned tile_w, unsigned tile_h,
    unsigned tensor_w, unsigned tensor_h,
    unsigned long long row_stride_elems,
    unsigned pad_interval_code, unsigned pad_enable) {
  unsigned long long ga = (unsigned long long)(uintptr_t)gaddr;
  u32x4 g0;
  g0[0] = 1u;                                   // count=1 (user mode)
  g0[1] = lds_off;                              // LDS byte address
  g0[2] = (unsigned)ga;                         // global_addr[31:0]
  g0[3] = (unsigned)((ga >> 32) & 0x1FFFFFFULL) | (2u << 30);  // [56:32] | type=2
  i32x8 g1;
  g1[0] = (int)((2u << 16) | (pad_enable << 20) | (pad_interval_code << 22));
  g1[1] = (int)((tensor_w & 0xFFFFu) << 16);
  g1[2] = (int)((tensor_w >> 16) | ((tensor_h & 0xFFFFu) << 16));
  g1[3] = (int)((tensor_h >> 16) | ((tile_w & 0xFFFFu) << 16));
  g1[4] = (int)(tile_h & 0xFFFFu);              // tile_dim1 ; tile_dim2 = 0
  g1[5] = (int)(unsigned)(row_stride_elems & 0xFFFFFFFFULL);
  g1[6] = (int)(unsigned)((row_stride_elems >> 32) & 0xFFFFULL);
  g1[7] = 0;
  i32x4 z4 = {0, 0, 0, 0};                      // 2D tensor: groups 2/3 unused
  i32x8 z8 = {0, 0, 0, 0, 0, 0, 0, 0};
  __builtin_amdgcn_tensor_load_to_lds(g0, g1, z4, z4, z8, 0);
}

// ======================= rmsnorm =======================
__global__ __launch_bounds__(256) void rmsnorm_kernel(
    const float* __restrict__ x, const float* __restrict__ w,
    float* __restrict__ y, int D) {
  __shared__ float red[256];
  const int t = blockIdx.x, tid = threadIdx.x;
  const float* xr = x + (size_t)t * D;
  float ss = 0.f;
  for (int d = tid; d < D; d += 256) { float v = xr[d]; ss += v * v; }
  red[tid] = ss; __syncthreads();
  for (int s = 128; s > 0; s >>= 1) {
    if (tid < s) red[tid] += red[tid + s];
    __syncthreads();
  }
  const float rinv = rsqrtf(red[0] / (float)D + 1e-6f);
  float* yr = y + (size_t)t * D;
  for (int d = tid; d < D; d += 256) yr[d] = xr[d] * rinv * w[d];
}

// ================= embedding gather + rmsnorm =================
__global__ __launch_bounds__(256) void embed_rmsnorm_kernel(
    const int* __restrict__ tok, const float* __restrict__ emb,
    const float* __restrict__ w, float* __restrict__ y, int D) {
  __shared__ float red[256];
  const int t = blockIdx.x, tid = threadIdx.x;
  const float* xr = emb + (size_t)tok[t] * D;
  float ss = 0.f;
  for (int d = tid; d < D; d += 256) { float v = xr[d]; ss += v * v; }
  red[tid] = ss; __syncthreads();
  for (int s = 128; s > 0; s >>= 1) {
    if (tid < s) red[tid] += red[tid + s];
    __syncthreads();
  }
  const float rinv = rsqrtf(red[0] / (float)D + 1e-6f);
  float* yr = y + (size_t)t * D;
  for (int d = tid; d < D; d += 256) yr[d] = xr[d] * rinv * w[d];
}

// ======================= WMMA GEMM with TDM staging =======================
// C[M,N] (+)= A[M,K] @ B[K,N]. fp32 tiles DMA'd Global->LDS by the Tensor Data
// Mover (double-buffered, TENSORcnt-tracked), converted f32->bf16 at fragment
// load, math on v_wmma_f32_16x16x32_bf16. 256 threads = 8 waves, 2(M)x4(N)
// wave grid, 64x32 per wave = 4x2 fragments.
#define BM 128
#define BN 128
#define BK 32
#define A_STRIDE 33    // TDM pads 1 DW per 32 DW  -> padded row stride
#define B_STRIDE 129   // TDM pads 1 DW per 128 DW -> padded row stride
#define EPI_STORE 0
#define EPI_ADD   1
#define EPI_BIAS  2
#define EPI_RSADD 3

template <int EPI>
__global__ __launch_bounds__(256) void gemm_bf16_wmma(
    const float* __restrict__ A, const float* __restrict__ Bm,
    float* __restrict__ C, int M, int N, int K,
    const float* __restrict__ bias,       // EPI_BIAS
    const float* __restrict__ rowscale,   // EPI_RSADD
    int rs_stride) {
  __shared__ float Asf[2][BM * A_STRIDE];   // [m][k], padded
  __shared__ float Bsf[2][BK * B_STRIDE];   // [k][n], padded (memory order)

  const int tid = threadIdx.x;
  const int wave = tid >> 5, lane = tid & 31;
  const int hf = lane >> 4, l16 = lane & 15;
  const int wm = wave & 1, wn = wave >> 1;
  const int m0 = blockIdx.y * BM, n0 = blockIdx.x * BN;

  const unsigned ldsA[2] = {(unsigned)(uintptr_t)&Asf[0][0],
                            (unsigned)(uintptr_t)&Asf[1][0]};
  const unsigned ldsB[2] = {(unsigned)(uintptr_t)&Bsf[0][0],
                            (unsigned)(uintptr_t)&Bsf[1][0]};

  v8f acc[4][2];
#pragma unroll
  for (int i = 0; i < 4; ++i)
#pragma unroll
    for (int j = 0; j < 2; ++j) acc[i][j] = (v8f){0,0,0,0,0,0,0,0};

  const int nsteps = K / BK;
  if (tid < 32) {  // wave 0 drives the DMA engine
    tdm_load_2d_f32(A + (size_t)m0 * K, ldsA[0], BK, BM,
                    K, M - m0, (unsigned long long)K, /*pad 32DW*/4, 1);
    tdm_load_2d_f32(Bm + n0, ldsB[0], BN, BK,
                    N - n0, K, (unsigned long long)N, /*pad 128DW*/6, 1);
  }

  for (int s = 0; s < nsteps; ++s) {
    if (tid < 32) __builtin_amdgcn_s_wait_tensorcnt(0);
    __syncthreads();  // tiles for step s are visible to all waves
    const int buf = s & 1;
    if (tid < 32 && s + 1 < nsteps) {  // overlap next DMA with current math
      const int k1 = (s + 1) * BK;
      tdm_load_2d_f32(A + (size_t)m0 * K + k1, ldsA[buf ^ 1], BK, BM,
                      K - k1, M - m0, (unsigned long long)K, 4, 1);
      tdm_load_2d_f32(Bm + (size_t)k1 * N + n0, ldsB[buf ^ 1], BN, BK,
                      N - n0, K - k1, (unsigned long long)N, 6, 1);
    }
    const float* Ab = &Asf[buf][0];
    const float* Bb = &Bsf[buf][0];

    // B fragments (contiguous k order, same as A: k = hf*16 + 2v + p, n = lane&15)
    v16bf bfrag[2];
#pragma unroll
    for (int nf = 0; nf < 2; ++nf) {
      const int nb = wn * 32 + nf * 16 + l16;
#pragma unroll
      for (int v = 0; v < 8; ++v) {
        const int kk = hf * 16 + v * 2;
        bfrag[nf][2 * v]     = f2bf(Bb[kk * B_STRIDE + nb]);
        bfrag[nf][2 * v + 1] = f2bf(Bb[(kk + 1) * B_STRIDE + nb]);
      }
    }
#pragma unroll
    for (int mf = 0; mf < 4; ++mf) {
      const int mb = wm * 64 + mf * 16 + l16;
      v16bf afrag;  // contiguous k order: k = hf*16 + s, same as B
#pragma unroll
      for (int v = 0; v < 8; ++v) {
        const int kk = hf * 16 + v * 2;
        afrag[2 * v]     = f2bf(Ab[mb * A_STRIDE + kk]);
        afrag[2 * v + 1] = f2bf(Ab[mb * A_STRIDE + kk + 1]);
      }
#pragma unroll
      for (int nf = 0; nf < 2; ++nf)
        acc[mf][nf] = wmma_bf16(afrag, bfrag[nf], acc[mf][nf]);
    }
    __syncthreads();  // all waves done with buf before it is refilled
  }

  // epilogue (C/D layout: m = vgpr + 8*hf, n = lane&15)
#pragma unroll
  for (int mf = 0; mf < 4; ++mf) {
#pragma unroll
    for (int nf = 0; nf < 2; ++nf) {
#pragma unroll
      for (int v = 0; v < 8; ++v) {
        const int row = m0 + wm * 64 + mf * 16 + hf * 8 + v;
        const int col = n0 + wn * 32 + nf * 16 + l16;
        const size_t idx = (size_t)row * N + col;
        float val = acc[mf][nf][v];
        if (EPI == EPI_STORE)      C[idx] = val;
        else if (EPI == EPI_ADD)   C[idx] = C[idx] + val;
        else if (EPI == EPI_BIAS)  C[idx] = val + bias[col];
        else /* EPI_RSADD */       C[idx] = C[idx] + rowscale[(size_t)row * rs_stride] * val;
      }
    }
  }
}

// ================= fused attention (per b, head, 64-row q tile) =================
// Keeps the full 64x1024 fp32 score stripe resident in the 320KB WGP LDS.
__global__ __launch_bounds__(256) void attention_kernel(
    const float* __restrict__ q, const float* __restrict__ k,
    const float* __restrict__ v, float* __restrict__ o, int Slen) {
  extern __shared__ char smem[];
  float*  Sc  = (float*)smem;              // [64][1024] fp32   (256 KB)
  float*  red = Sc + 64 * 1024;            // [64][4]           (1 KB)
  __bf16* Qs  = (__bf16*)(red + 64 * 4);   // [64][128] bf16    (16 KB)

  const int b = blockIdx.z, hh = blockIdx.y, qt = blockIdx.x;
  const size_t rs = (size_t)NH * HD;  // 2048 row stride
  const float* qb = q + ((size_t)b * Slen + qt * 64) * rs + hh * HD;
  const float* kb = k + (size_t)b * Slen * rs + hh * HD;
  const float* vb = v + (size_t)b * Slen * rs + hh * HD;
  float*       ob = o + ((size_t)b * Slen + qt * 64) * rs + hh * HD;

  const int tid = threadIdx.x;
  const int wave = tid >> 5, lane = tid & 31;
  const int hf = lane >> 4, l16 = lane & 15;

  // Q tile -> LDS bf16
  for (int i = tid; i < 64 * 128; i += 256) {
    int r = i >> 7, c = i & 127;
    Qs[r * 128 + c] = f2bf(qb[(size_t)r * rs + c]);
  }
  __syncthreads();

  const float scale = 0.08838834764831845f;  // 1/sqrt(128)

  // ---- scores = Q @ K^T : M=64, N=Slen, Kdim=128. K row-major == Bt[n][k]. ----
  for (int j = 0; j < Slen / 128; ++j) {
    const int nfr = wave + 8 * j;
    const int nbase = nfr * 16;
    v8f acc[4];
#pragma unroll
    for (int mf = 0; mf < 4; ++mf) acc[mf] = (v8f){0,0,0,0,0,0,0,0};
    for (int kk = 0; kk < 4; ++kk) {
      v16bf bf;
      const float* kp = kb + (size_t)(nbase + l16) * rs + kk * 32 + hf * 16;
#pragma unroll
      for (int vv = 0; vv < 8; ++vv) {
        bf[2 * vv]     = f2bf(kp[2 * vv]);
        bf[2 * vv + 1] = f2bf(kp[2 * vv + 1]);
      }
#pragma unroll
      for (int mf = 0; mf < 4; ++mf) {
        const int mrow = mf * 16 + l16;
        v16bf af;
#pragma unroll
        for (int vv = 0; vv < 8; ++vv) {
          const int kd = kk * 32 + hf * 16 + vv * 2;
          af[2 * vv]     = Qs[mrow * 128 + kd];
          af[2 * vv + 1] = Qs[mrow * 128 + kd + 1];
        }
        acc[mf] = wmma_bf16(af, bf, acc[mf]);
      }
    }
#pragma unroll
    for (int mf = 0; mf < 4; ++mf)
#pragma unroll
      for (int vv = 0; vv < 8; ++vv)
        Sc[(mf * 16 + hf * 8 + vv) * 1024 + nbase + l16] = acc[mf][vv] * scale;
  }
  __syncthreads();

  // ---- softmax over each of 64 rows; 4 threads per row ----
  {
    const int row = tid >> 2, sub = tid & 3;
    float* r = Sc + (size_t)row * 1024 + sub * 256;
    float mx = -INFINITY;
    for (int i = 0; i < 256; ++i) mx = fmaxf(mx, r[i]);
    red[row * 4 + sub] = mx; __syncthreads();
    mx = fmaxf(fmaxf(red[row * 4 + 0], red[row * 4 + 1]),
               fmaxf(red[row * 4 + 2], red[row * 4 + 3]));
    __syncthreads();
    float sm = 0.f;
    for (int i = 0; i < 256; ++i) { float e = __expf(r[i] - mx); r[i] = e; sm += e; }
    red[row * 4 + sub] = sm; __syncthreads();
    sm = red[row * 4 + 0] + red[row * 4 + 1] + red[row * 4 + 2] + red[row * 4 + 3];
    const float inv = 1.f / sm;
    for (int i = 0; i < 256; ++i) r[i] *= inv;
  }
  __syncthreads();

  // ---- O = P @ V : M=64, N=128, Kdim=Slen. Each wave owns one 16-col frag. ----
  {
    const int nbase = wave * 16;
    v8f oacc[4];
#pragma unroll
    for (int mf = 0; mf < 4; ++mf) oacc[mf] = (v8f){0,0,0,0,0,0,0,0};
    for (int kk = 0; kk < Slen / 32; ++kk) {
      v16bf bf;
#pragma unroll
      for (int vv = 0; vv < 8; ++vv) {
        const int s1 = kk * 32 + hf * 16 + 2 * vv;
        bf[2 * vv]     = f2bf(vb[(size_t)s1 * rs + nbase + l16]);
        bf[2 * vv + 1] = f2bf(vb[(size_t)(s1 + 1) * rs + nbase + l16]);
      }
#pragma unroll
      for (int mf = 0; mf < 4; ++mf) {
        const int mrow = mf * 16 + l16;
        v16bf af;  // P read from LDS fp32, cast to bf16 at fragment load
#pragma unroll
        for (int vv = 0; vv < 8; ++vv) {
          const int ks = kk * 32 + hf * 16 + vv * 2;
          af[2 * vv]     = f2bf(Sc[mrow * 1024 + ks]);
          af[2 * vv + 1] = f2bf(Sc[mrow * 1024 + ks + 1]);
        }
        oacc[mf] = wmma_bf16(af, bf, oacc[mf]);
      }
    }
#pragma unroll
    for (int mf = 0; mf < 4; ++mf)
#pragma unroll
      for (int vv = 0; vv < 8; ++vv)
        ob[(size_t)(mf * 16 + hf * 8 + vv) * rs + nbase + l16] = oacc[mf][vv];
  }
}

// ================= MoE gate: softmax(x@Wg) -> top-2 normalized =================
__global__ __launch_bounds__(128) void gate_topk_kernel(
    const float* __restrict__ xn, const float* __restrict__ Wg,
    float* __restrict__ gw, int D) {
  __shared__ float part[128][NE];
  const int t = blockIdx.x, tid = threadIdx.x;
  float a[NE] = {0.f, 0.f, 0.f, 0.f};
  const float* xr = xn + (size_t)t * D;
  for (int d = tid; d < D; d += 128) {
    const float x = xr[d];
#pragma unroll
    for (int e = 0; e < NE; ++e) a[e] += x * Wg[(size_t)d * NE + e];
  }
#pragma unroll
  for (int e = 0; e < NE; ++e) part[tid][e] = a[e];
  __syncthreads();
  for (int s = 64; s > 0; s >>= 1) {
    if (tid < s)
#pragma unroll
      for (int e = 0; e < NE; ++e) part[tid][e] += part[tid + s][e];
    __syncthreads();
  }
  if (tid == 0) {
    float g[NE];
    float mx = part[0][0];
#pragma unroll
    for (int e = 1; e < NE; ++e) mx = fmaxf(mx, part[0][e]);
    float sm = 0.f;
#pragma unroll
    for (int e = 0; e < NE; ++e) { g[e] = __expf(part[0][e] - mx); sm += g[e]; }
#pragma unroll
    for (int e = 0; e < NE; ++e) g[e] /= sm;
    int i0 = 0;
#pragma unroll
    for (int e = 1; e < NE; ++e) if (g[e] > g[i0]) i0 = e;
    int i1 = (i0 == 0) ? 1 : 0;
#pragma unroll
    for (int e = 0; e < NE; ++e) if (e != i0 && g[e] > g[i1]) i1 = e;
    const float norm = g[i0] + g[i1];
#pragma unroll
    for (int e = 0; e < NE; ++e)
      gw[(size_t)t * NE + e] = (e == i0 || e == i1) ? (g[e] / norm) : 0.f;
  }
}

// ================= silu(h1) * h3 =================
__global__ __launch_bounds__(256) void silu_mul_kernel(
    const float* __restrict__ h1, const float* __restrict__ h3,
    float* __restrict__ hg, int n) {
  const int i = blockIdx.x * 256 + threadIdx.x;
  if (i < n) {
    const float x = h1[i];
    hg[i] = (x / (1.f + __expf(-x))) * h3[i];
  }
}

// ================= host-side launcher =================
static void gemm(hipStream_t s, const float* A, const float* B, float* C,
                 int M, int N, int K, int epi,
                 const float* bias, const float* rowscale, int rs_stride) {
  dim3 grid(N / BN, M / BM), block(256);
  switch (epi) {
    case EPI_STORE:
      gemm_bf16_wmma<EPI_STORE><<<grid, block, 0, s>>>(A, B, C, M, N, K, bias, rowscale, rs_stride);
      break;
    case EPI_ADD:
      gemm_bf16_wmma<EPI_ADD><<<grid, block, 0, s>>>(A, B, C, M, N, K, bias, rowscale, rs_stride);
      break;
    case EPI_BIAS:
      gemm_bf16_wmma<EPI_BIAS><<<grid, block, 0, s>>>(A, B, C, M, N, K, bias, rowscale, rs_stride);
      break;
    default:
      gemm_bf16_wmma<EPI_RSADD><<<grid, block, 0, s>>>(A, B, C, M, N, K, bias, rowscale, rs_stride);
      break;
  }
}

extern "C" void kernel_launch(void* const* d_in, const int* in_sizes, int n_in,
                              void* d_out, int out_size, void* d_ws, size_t ws_size,
                              hipStream_t stream) {
  const int*   tok    = (const int*)d_in[0];
  const float* emb    = (const float*)d_in[1];
  const float* enw    = (const float*)d_in[2];
  const float* normw  = (const float*)d_in[3];   // [DEPTH, DIM]
  const float* Wq     = (const float*)d_in[4];   // [DEPTH, DIM, 2048]
  const float* Wdkv   = (const float*)d_in[5];   // [DEPTH, DIM, 512]
  const float* Wuk    = (const float*)d_in[6];   // [DEPTH, 512, 2048]
  const float* Wuv    = (const float*)d_in[7];   // [DEPTH, 512, 2048]
  const float* Wo     = (const float*)d_in[8];   // [DEPTH, 2048, DIM]
  const float* Wg     = (const float*)d_in[9];   // [DEPTH, DIM, 4]
  const float* W1     = (const float*)d_in[10];  // [DEPTH, E, DIM, DFF]
  const float* W3     = (const float*)d_in[11];
  const float* W2     = (const float*)d_in[12];  // [DEPTH, E, DFF, DIM]
  const float* hnw    = (const float*)d_in[13];
  const float* headW  = (const float*)d_in[14];  // [DIM, V]
  const float* headb  = (const float*)d_in[15];
  float* out = (float*)d_out;

  // workspace partition (floats)
  float* ws = (float*)d_ws;
  float* h  = ws;                 // [T, DIM]
  float* xn = h  + (size_t)TT * DIM;
  float* qb = xn + (size_t)TT * DIM;   // [T, H*HD]
  float* cb = qb + (size_t)TT * DIM;   // [T, R]
  float* kb = cb + (size_t)TT * RLAT;  // [T, H*HD]
  float* vb = kb + (size_t)TT * DIM;   // [T, H*HD]
  float* ao = vb + (size_t)TT * DIM;   // [T, H*HD]
  float* h1 = ao + (size_t)TT * DIM;   // [T, DFF]
  float* h3 = h1 + (size_t)TT * DFF;
  float* hg = h3 + (size_t)TT * DFF;
  float* gw = hg + (size_t)TT * DFF;   // [T, E]
  (void)in_sizes; (void)n_in; (void)out_size; (void)ws_size;

  // 1. embed + rmsnorm
  embed_rmsnorm_kernel<<<TT, 256, 0, stream>>>(tok, emb, enw, h, DIM);

  const size_t attn_lds = (size_t)64 * 1024 * 4 + 64 * 4 * 4 + (size_t)64 * 128 * 2;

  for (int l = 0; l < DEPTH; ++l) {
    const float* nw = normw + (size_t)l * DIM;
    // ---- MLA ----
    rmsnorm_kernel<<<TT, 256, 0, stream>>>(h, nw, xn, DIM);
    gemm(stream, xn, Wq   + (size_t)l * DIM * DIM,  qb, TT, DIM,  DIM,  EPI_STORE, nullptr, nullptr, 0);
    gemm(stream, xn, Wdkv + (size_t)l * DIM * RLAT, cb, TT, RLAT, DIM,  EPI_STORE, nullptr, nullptr, 0);
    gemm(stream, cb, Wuk  + (size_t)l * RLAT * DIM, kb, TT, DIM,  RLAT, EPI_STORE, nullptr, nullptr, 0);
    gemm(stream, cb, Wuv  + (size_t)l * RLAT * DIM, vb, TT, DIM,  RLAT, EPI_STORE, nullptr, nullptr, 0);
    attention_kernel<<<dim3(SS / 64, NH, BB), 256, attn_lds, stream>>>(qb, kb, vb, ao, SS);
    gemm(stream, ao, Wo + (size_t)l * DIM * DIM, h, TT, DIM, DIM, EPI_ADD, nullptr, nullptr, 0);

    // ---- MoE (same norm weight reused, as in reference) ----
    rmsnorm_kernel<<<TT, 256, 0, stream>>>(h, nw, xn, DIM);
    gate_topk_kernel<<<TT, 128, 0, stream>>>(xn, Wg + (size_t)l * DIM * NE, gw, DIM);
    for (int e = 0; e < NE; ++e) {
      const size_t woff = ((size_t)l * NE + e) * (size_t)DIM * DFF;
      gemm(stream, xn, W1 + woff, h1, TT, DFF, DIM, EPI_STORE, nullptr, nullptr, 0);
      gemm(stream, xn, W3 + woff, h3, TT, DFF, DIM, EPI_STORE, nullptr, nullptr, 0);
      silu_mul_kernel<<<(TT * DFF + 255) / 256, 256, 0, stream>>>(h1, h3, hg, TT * DFF);
      gemm(stream, hg, W2 + ((size_t)l * NE + e) * (size_t)DFF * DIM, h,
           TT, DIM, DFF, EPI_RSADD, nullptr, gw + e, NE);
    }
  }

  // 3. head
  rmsnorm_kernel<<<TT, 256, 0, stream>>>(h, hnw, xn, DIM);
  gemm(stream, xn, headW, out, TT, VOC, DIM, EPI_BIAS, headb, nullptr, 0);
}
